// ParallelSequenceMoEHead_62156766708379
// MI455X (gfx1250) — compile-verified, matrix operand order checked by reading, not run on hardware
//
#include <hip/hip_runtime.h>
#include <hip/hip_bf16.h>
#include <stdint.h>

// MoE head for MI455X (gfx1250, wave32).
// Top-2 routing first; only the 2 selected experts per batch item are run
// (4x FLOP savings vs reference's dense all-expert compute). GEMMs use
// v_wmma_f32_16x16x32_bf16 with f32 accumulate, double-buffered LDS tiles,
// and GLOBAL_LOAD_ASYNC_TO_LDS_B128 for the (already-bf16) activation
// operand of GEMM2. Workspace: 256 B header + 16 MB bf16 hidden.

typedef __bf16 bf16_t;
typedef __attribute__((ext_vector_type(16))) __bf16 v16bf;
typedef __attribute__((ext_vector_type(8)))  float  v8f;

#define NEXP 8
#define NE   1024   // n_embd == text_dim == full
#define BSZ  8
#define SEQL 512

#define BM 128
#define BN 128
#define BK 32
#define LDA 40      // padded LDS stride (shorts); multiple of 8 for b128 ops
#define LDB 40
#define NT  (NE / BK)   // 32 k-steps per GEMM

struct Bits256 { uint4 a, b; };   // 32 bytes -> bit_cast to v16bf

__device__ __forceinline__ uint16_t f2bf(float f) {
  uint32_t u = __builtin_bit_cast(uint32_t, f);
  u += 0x7FFFu + ((u >> 16) & 1u);        // round-to-nearest-even
  return (uint16_t)(u >> 16);
}

// ---------------------------------------------------------------------------
// Router: logits, softmax, top-2, gates, mask, aux loss. One small block.
// ---------------------------------------------------------------------------
__global__ __launch_bounds__(64) void router_kernel(
    const float* __restrict__ dec, const float* __restrict__ rw,
    const float* __restrict__ rb,
    float* __restrict__ out_aux, float* __restrict__ out_mask,
    float* __restrict__ out_logits,
    int* __restrict__ ws_idx, float* __restrict__ ws_w) {
  __shared__ float probs[BSZ][NEXP];
  __shared__ float mask[BSZ][NEXP];
  const int t = threadIdx.x;
  const int b = t >> 3, e = t & 7;

  float acc = rb[e];
  const float* dv = dec + b * NE;
  const float* wv = rw + e * NE;
  for (int i = 0; i < NE; i += 4) {
    float4 d = *(const float4*)(dv + i);
    float4 w = *(const float4*)(wv + i);
    acc += d.x * w.x + d.y * w.y + d.z * w.z + d.w * w.w;
  }
  out_logits[b * NEXP + e] = acc;
  probs[b][e] = acc;
  __syncthreads();

  if (t < BSZ) {
    const int bb = t;
    float mx = probs[bb][0];
    for (int i = 1; i < NEXP; ++i) mx = fmaxf(mx, probs[bb][i]);
    float p[NEXP], s = 0.f;
    for (int i = 0; i < NEXP; ++i) { p[i] = __expf(probs[bb][i] - mx); s += p[i]; }
    for (int i = 0; i < NEXP; ++i) p[i] /= s;
    int i0 = 0;
    for (int i = 1; i < NEXP; ++i) if (p[i] > p[i0]) i0 = i;
    int i1 = (i0 == 0) ? 1 : 0;
    for (int i = 0; i < NEXP; ++i) if (i != i0 && p[i] > p[i1]) i1 = i;
    const float den = p[i0] + p[i1];
    ws_idx[bb * 2 + 0] = i0;  ws_idx[bb * 2 + 1] = i1;
    ws_w[bb * 2 + 0] = p[i0] / den;  ws_w[bb * 2 + 1] = p[i1] / den;
    for (int i = 0; i < NEXP; ++i) {
      mask[bb][i] = (i == i0 || i == i1) ? 1.f : 0.f;
      probs[bb][i] = p[i];
      out_mask[bb * NEXP + i] = mask[bb][i];
    }
  }
  __syncthreads();

  if (t == 0) {
    float loss = 0.f;
    for (int ee = 0; ee < NEXP; ++ee) {
      float mp = 0.f, mm = 0.f;
      for (int bb = 0; bb < BSZ; ++bb) { mp += probs[bb][ee]; mm += mask[bb][ee]; }
      loss += (mp / BSZ) * (mm / BSZ);
    }
    out_aux[0] = (float)NEXP * loss;
  }
}

// ---------------------------------------------------------------------------
// WMMA tile machinery: 256 threads (8 waves), block tile 128x128x32, wave
// grid 2(M) x 4(N), each wave 64x32 = 4x2 fragments of 16x16.
// 16-bit A/B fragment layout per CDNA5 ISA: lane L -> row=L&15, half=L>>4;
// dwords 0..3 hold K = half*8 .. half*8+7, dwords 4..7 hold +16 -> two
// contiguous ds_load_b128 per fragment.
// ---------------------------------------------------------------------------
__device__ __forceinline__ v16bf load_frag(const uint16_t* lds, int row,
                                           int ld, int lhi) {
  const uint4* p = (const uint4*)(lds + row * ld + lhi * 8);
  Bits256 u;
  u.a = p[0];
  u.b = p[2];   // +32 bytes -> K += 16
  return __builtin_bit_cast(v16bf, u);
}

__device__ __forceinline__ void wmma_tile(const uint16_t* sa, const uint16_t* sb,
                                          int wm, int wn, int lrow, int lhi,
                                          v8f (&acc)[4][2]) {
  v16bf bfrag[2];
#pragma unroll
  for (int nf = 0; nf < 2; ++nf)
    bfrag[nf] = load_frag(sb, wn + nf * 16 + lrow, LDB, lhi);
#pragma unroll
  for (int mf = 0; mf < 4; ++mf) {
    v16bf afrag = load_frag(sa, wm + mf * 16 + lrow, LDA, lhi);
#pragma unroll
    for (int nf = 0; nf < 2; ++nf)
      acc[mf][nf] = __builtin_amdgcn_wmma_f32_16x16x32_bf16(
          false, afrag, false, bfrag[nf], (short)0, acc[mf][nf], false, false);
  }
}

// Stage a 32x128 f32 weight tile, converted to bf16, transposed: lds[n][k].
__device__ __forceinline__ void stage_b_f32(const float* we, int k0, int colBase,
                                            int tid, uint16_t* sb) {
#pragma unroll
  for (int c = 0; c < 4; ++c) {
    const int lin = c * 256 + tid;
    const int kr = lin >> 5;
    const int c4 = (lin & 31) << 2;
    float4 f = *(const float4*)(we + (size_t)(k0 + kr) * NE + colBase + c4);
    sb[(c4 + 0) * LDB + kr] = f2bf(f.x);
    sb[(c4 + 1) * LDB + kr] = f2bf(f.y);
    sb[(c4 + 2) * LDB + kr] = f2bf(f.z);
    sb[(c4 + 3) * LDB + kr] = f2bf(f.w);
  }
}

// Stage a 128x32 f32 activation tile, converted to bf16: lds[m][k].
__device__ __forceinline__ void stage_a_f32(const float* xb, int k0, int rowBase,
                                            int tid, uint16_t* sa) {
#pragma unroll
  for (int c = 0; c < 4; ++c) {
    const int lin = c * 256 + tid;
    const int r = lin >> 3;
    const int c4 = (lin & 7) << 2;
    float4 f = *(const float4*)(xb + (size_t)(rowBase + r) * NE + k0 + c4);
    uint16_t* d = &sa[r * LDA + c4];
    d[0] = f2bf(f.x); d[1] = f2bf(f.y); d[2] = f2bf(f.z); d[3] = f2bf(f.w);
  }
}

// Stage a 128x32 bf16 activation tile straight from global into LDS via the
// gfx1250 async-to-LDS path (no VGPR round trip). Publisher must
// s_wait_asynccnt 0 before the barrier.
__device__ __forceinline__ void stage_a_async(const uint16_t* ha, int k0,
                                              int rowBase, int tid,
                                              uint16_t* sa) {
#pragma unroll
  for (int c = 0; c < 2; ++c) {
    const int lin = c * 256 + tid;        // 0..511 b128 chunks
    const int r = lin >> 2;               // 4 x 16B per 32-short row
    const int cq = (lin & 3) << 3;        // short offset, 16B granules
    const uint16_t* src = ha + (size_t)(rowBase + r) * NE + k0 + cq;
    const uint32_t dst = (uint32_t)(size_t)(sa + r * LDA + cq);  // LDS byte offset
    asm volatile("global_load_async_to_lds_b128 %0, %1, off"
                 :: "v"(dst), "v"(src) : "memory");
  }
}

__device__ __forceinline__ void wait_async_lds() {
  asm volatile("s_wait_asynccnt 0x0" ::: "memory");
}

// ---------------------------------------------------------------------------
// GEMM1: hidden[b,slot] = relu(x[b] @ w1[e] + b1[e])  -> bf16 in ws.
// grid: (N tiles=8, M tiles=4, b*2+slot=16). Double-buffered LDS.
// ---------------------------------------------------------------------------
__global__ __launch_bounds__(256) void gemm1_kernel(
    const float* __restrict__ x, const float* __restrict__ w1,
    const float* __restrict__ b1, const int* __restrict__ ws_idx,
    uint16_t* __restrict__ hidden) {
  __shared__ __align__(16) uint16_t lds_a[2][BM * LDA];
  __shared__ __align__(16) uint16_t lds_b[2][BN * LDB];

  const int tid = threadIdx.x;
  const int bz = blockIdx.z;                 // b*2 + slot
  const int b = bz >> 1;
  const int ex = ws_idx[bz];
  const int rowBase = blockIdx.y * BM;       // S
  const int colBase = blockIdx.x * BN;       // H
  const float* xb = x + (size_t)b * SEQL * NE;
  const float* we = w1 + (size_t)ex * NE * NE;

  const int lane = tid & 31;
  const int wave = tid >> 5;
  const int wm = (wave & 1) * 64;
  const int wn = (wave >> 1) * 32;
  const int lrow = lane & 15;
  const int lhi = lane >> 4;

  const v8f vzero = {0.f, 0.f, 0.f, 0.f, 0.f, 0.f, 0.f, 0.f};
  v8f acc[4][2];
#pragma unroll
  for (int i = 0; i < 4; ++i)
#pragma unroll
    for (int j = 0; j < 2; ++j) acc[i][j] = vzero;

  stage_a_f32(xb, 0, rowBase, tid, lds_a[0]);
  stage_b_f32(we, 0, colBase, tid, lds_b[0]);
  __syncthreads();

  for (int t = 0; t < NT; ++t) {
    const int cur = t & 1;
    if (t + 1 < NT) {
      stage_a_f32(xb, (t + 1) * BK, rowBase, tid, lds_a[cur ^ 1]);
      stage_b_f32(we, (t + 1) * BK, colBase, tid, lds_b[cur ^ 1]);
    }
    wmma_tile(lds_a[cur], lds_b[cur], wm, wn, lrow, lhi, acc);
    __syncthreads();
  }

  uint16_t* hb = hidden + (size_t)bz * SEQL * NE;
  const float* bias = b1 + (size_t)ex * NE;
#pragma unroll
  for (int mf = 0; mf < 4; ++mf)
#pragma unroll
    for (int nf = 0; nf < 2; ++nf) {
      const int h = colBase + wn + nf * 16 + lrow;
      const float bv = bias[h];
#pragma unroll
      for (int v = 0; v < 8; ++v) {
        const int s = rowBase + wm + mf * 16 + lhi * 8 + v;
        hb[(size_t)s * NE + h] = f2bf(fmaxf(acc[mf][nf][v] + bv, 0.f));
      }
    }
}

// ---------------------------------------------------------------------------
// GEMM2: out[b] = sum_slot gate_s * (hidden[b,slot] @ w2[e_s]) + gated b2.
// Single accumulator across both slots: run slot0 unscaled, rescale acc by
// g0/g1 at the slot boundary, multiply by g1 in the epilogue.
// grid: (N tiles=8, M tiles=4, b=8). Double-buffered; A via async-to-LDS.
// ---------------------------------------------------------------------------
__global__ __launch_bounds__(256) void gemm2_kernel(
    const uint16_t* __restrict__ hidden, const float* __restrict__ w2,
    const float* __restrict__ b2, const int* __restrict__ ws_idx,
    const float* __restrict__ ws_w, float* __restrict__ out) {
  __shared__ __align__(16) uint16_t lds_a[2][BM * LDA];
  __shared__ __align__(16) uint16_t lds_b[2][BN * LDB];

  const int tid = threadIdx.x;
  const int b = blockIdx.z;
  const int rowBase = blockIdx.y * BM;       // S
  const int colBase = blockIdx.x * BN;       // F

  const int e0 = ws_idx[b * 2 + 0], e1 = ws_idx[b * 2 + 1];
  const float g0 = ws_w[b * 2 + 0], g1 = ws_w[b * 2 + 1];
  const float ratio = g0 / g1;               // softmax gates: g1 > 0
  const uint16_t* haS[2] = {hidden + (size_t)(b * 2 + 0) * SEQL * NE,
                            hidden + (size_t)(b * 2 + 1) * SEQL * NE};
  const float* weS[2] = {w2 + (size_t)e0 * NE * NE, w2 + (size_t)e1 * NE * NE};

  const int lane = tid & 31;
  const int wave = tid >> 5;
  const int wm = (wave & 1) * 64;
  const int wn = (wave >> 1) * 32;
  const int lrow = lane & 15;
  const int lhi = lane >> 4;

  const v8f vzero = {0.f, 0.f, 0.f, 0.f, 0.f, 0.f, 0.f, 0.f};
  v8f acc[4][2];
#pragma unroll
  for (int i = 0; i < 4; ++i)
#pragma unroll
    for (int j = 0; j < 2; ++j) acc[i][j] = vzero;

  const int NS = 2 * NT;                     // 64 combined k-steps
  stage_a_async(haS[0], 0, rowBase, tid, lds_a[0]);
  stage_b_f32(weS[0], 0, colBase, tid, lds_b[0]);
  wait_async_lds();
  __syncthreads();

  for (int s = 0; s < NS; ++s) {
    const int cur = s & 1;
    if (s + 1 < NS) {
      const int slot = (s + 1) >> 5;
      const int k0 = ((s + 1) & (NT - 1)) * BK;
      stage_a_async(haS[slot], k0, rowBase, tid, lds_a[cur ^ 1]);
      stage_b_f32(weS[slot], k0, colBase, tid, lds_b[cur ^ 1]);
    }
    wmma_tile(lds_a[cur], lds_b[cur], wm, wn, lrow, lhi, acc);
    if (s == NT - 1) {                       // slot boundary: acc *= g0/g1
#pragma unroll
      for (int i = 0; i < 4; ++i)
#pragma unroll
        for (int j = 0; j < 2; ++j)
#pragma unroll
          for (int v = 0; v < 8; ++v) acc[i][j][v] *= ratio;
    }
    wait_async_lds();
    __syncthreads();
  }

#pragma unroll
  for (int mf = 0; mf < 4; ++mf)
#pragma unroll
    for (int nf = 0; nf < 2; ++nf) {
      const int f = colBase + wn + nf * 16 + lrow;
      const float bv = g0 * b2[(size_t)e0 * NE + f] + g1 * b2[(size_t)e1 * NE + f];
#pragma unroll
      for (int v = 0; v < 8; ++v) {
        const int s = rowBase + wm + mf * 16 + lhi * 8 + v;
        out[((size_t)b * SEQL + s) * NE + f] = g1 * acc[mf][nf][v] + bv;
      }
    }
}

// ---------------------------------------------------------------------------
extern "C" void kernel_launch(void* const* d_in, const int* in_sizes, int n_in,
                              void* d_out, int out_size, void* d_ws, size_t ws_size,
                              hipStream_t stream) {
  const float* x   = (const float*)d_in[0];   // (8,512,1024)
  const float* dec = (const float*)d_in[1];   // (8,1024)
  const float* rw  = (const float*)d_in[2];   // (8,1024)
  const float* rb  = (const float*)d_in[3];   // (8,)
  const float* w1  = (const float*)d_in[4];   // (8,1024,1024)
  const float* w2  = (const float*)d_in[5];   // (8,1024,1024)
  const float* b1  = (const float*)d_in[6];   // (8,1,1024)
  const float* b2  = (const float*)d_in[7];   // (8,1,1024)

  float* out = (float*)d_out;
  float* out_final  = out;                                   // 8*512*1024
  float* out_aux    = out + (size_t)BSZ * SEQL * NE;         // 1
  float* out_mask   = out_aux + 1;                           // 64
  float* out_logits = out_mask + 64;                         // 64

  // ws layout: [0,64) int topk_idx[8][2]; [64,128) float topk_w[8][2];
  // [256, 256+16MB) bf16 hidden[8][2][512][1024]
  int*      ws_idx = (int*)d_ws;
  float*    ws_w   = (float*)((char*)d_ws + 64);
  uint16_t* hidden = (uint16_t*)((char*)d_ws + 256);

  router_kernel<<<dim3(1), dim3(64), 0, stream>>>(dec, rw, rb, out_aux,
                                                  out_mask, out_logits,
                                                  ws_idx, ws_w);

  dim3 blk(256);
  dim3 g1(NE / BN, SEQL / BM, BSZ * 2);   // 8 x 4 x 16
  gemm1_kernel<<<g1, blk, 0, stream>>>(x, w1, b1, ws_idx, hidden);

  dim3 g2(NE / BN, SEQL / BM, BSZ);       // 8 x 4 x 8
  gemm2_kernel<<<g2, blk, 0, stream>>>(hidden, w2, b2, ws_idx, ws_w, out_final);
}
